// FinancialGNNRAG_72834055405693
// MI455X (gfx1250) — matrix operand order repaired and checked by reading, not compiled, not run on hardware
//
#include <hip/hip_runtime.h>
#include <hip/hip_fp16.h>

typedef __attribute__((ext_vector_type(16))) _Float16 v16h;
typedef __attribute__((ext_vector_type(8)))  _Float16 v8h;
typedef __attribute__((ext_vector_type(4)))  _Float16 v4h;
typedef __attribute__((ext_vector_type(8)))  float    v8f;

#define NEG_SLOPE 0.2f

// ---------------------------------------------------------------- utilities

__global__ void k_fill_f32(float* p, float v, int n) {
    int i = blockIdx.x * blockDim.x + threadIdx.x;
    if (i < n) p[i] = v;
}

__global__ void k_fill_u32(unsigned* p, unsigned v, int n) {
    int i = blockIdx.x * blockDim.x + threadIdx.x;
    if (i < n) p[i] = v;
}

// vectorized f32 -> f16 (4 elems/thread), n % 4 == 0
__global__ void k_f32_to_f16_v4(const float* __restrict__ in, _Float16* __restrict__ out, int n4) {
    int i = blockIdx.x * blockDim.x + threadIdx.x;
    if (i >= n4) return;
    const float4 v = ((const float4*)in)[i];
    v4h o; o[0] = (_Float16)v.x; o[1] = (_Float16)v.y; o[2] = (_Float16)v.z; o[3] = (_Float16)v.w;
    ((v4h*)out)[i] = o;
}

// ------------------------------------------------------- layer-1 GEMM (K=3)

__global__ void k_gemm_small_k(const float* __restrict__ x, const float* __restrict__ W,
                               float* __restrict__ out, int n, int K, int Nout) {
    int i = blockIdx.x * blockDim.x + threadIdx.x;
    if (i >= n * Nout) return;
    int row = i / Nout, col = i % Nout;
    float s = 0.f;
    for (int k = 0; k < K; ++k) s += x[row * K + k] * W[k * Nout + col];
    out[i] = s;
}

// ----------------------------------------------- B pack: W[K,Nout] f32 ->
// strip-fragment-ordered f16:
//   Bp[((((g*nks + ks)*NT + t)*32 + lane)*16) + j]
// with lane={lo,hi}: K = ks*32 + (j>>3)*16 + hi*8 + (j&7), col = (g*NT + t)*16 + lo.
// A wave working on column-group g reads, per k-step, NT contiguous 512-half
// fragments at constant immediate offsets from one pointer.

__global__ void k_pack_B(const float* __restrict__ W, _Float16* __restrict__ Bp,
                         int K, int Nout, int NT) {
    int i = blockIdx.x * blockDim.x + threadIdx.x;
    if (i >= K * Nout) return;
    int nks  = K >> 5;
    int j    = i & 15;
    int lane = (i >> 4) & 31;
    int q    = i >> 9;                 // (g*nks + ks)*NT + t
    int t    = q % NT;
    int q2   = q / NT;                 // g*nks + ks
    int ks   = q2 % nks;
    int g    = q2 / nks;
    int lo = lane & 15, hi = lane >> 4;
    int k   = (ks << 5) + ((j >> 3) << 4) + (hi << 3) + (j & 7);
    int col = ((g * NT + t) << 4) + lo;
    Bp[i] = (_Float16)W[(size_t)k * Nout + col];
}

// ------------------------------------------------- WMMA GEMM (f16 -> f32)
// One wave computes a 16 x (16*NT) strip of D.  A: [M,K] f16 row-major
// (per-lane fragment = two contiguous 16B chunks), B: strip-packed (above),
// C: [M,Nout] f32.  M%16==0, K%32==0, Nout%(16*NT)==0.

template <int NT>
__global__ void k_wmma_gemm_f16(const _Float16* __restrict__ A, const _Float16* __restrict__ Bp,
                                float* __restrict__ C, int K, int Nout, int totalWaves) {
    int wave = (int)((blockIdx.x * blockDim.x + threadIdx.x) >> 5);
    if (wave >= totalWaves) return;          // wave-uniform: EXEC all-1 for WMMA
    int lane = threadIdx.x & 31;
    int lo = lane & 15, hi = lane >> 4;

    const int nks = K >> 5;                  // k-steps
    const int nColGroups = Nout / (16 * NT); // strips per row of tiles
    int tm = (wave / nColGroups) << 4;
    int g  = (wave % nColGroups);            // column group

    v8f acc[NT];
#pragma unroll
    for (int t = 0; t < NT; ++t) acc[t] = (v8f){};

    const _Float16* ap = A + (size_t)(tm + lo) * K + (hi << 3);
    const _Float16* bp = Bp + (size_t)g * nks * (NT * 512) + lane * 16;

    for (int ks = 0; ks < nks; ++ks) {
        v8h a0 = *(const v8h*)(ap);          // K = kk + hi*8 + 0..7
        v8h a1 = *(const v8h*)(ap + 16);     // K = kk + 16 + hi*8 + 0..7
        v16h a = __builtin_shufflevector(a0, a1, 0, 1, 2, 3, 4, 5, 6, 7,
                                         8, 9, 10, 11, 12, 13, 14, 15);
#pragma unroll
        for (int t = 0; t < NT; ++t) {
            const v16h b = *(const v16h*)(bp + t * 512);   // constant imm offsets
            acc[t] = __builtin_amdgcn_wmma_f32_16x16x32_f16(
                false, a, false, b, (short)0, acc[t], false, false);
        }
        ap += 32;            // next 32-wide K slice of this row
        bp += NT * 512;      // next k-step's NT fragments (contiguous)
    }
#pragma unroll
    for (int t = 0; t < NT; ++t) {
        float* Cb = C + (size_t)(tm + (hi << 3)) * Nout + ((g * NT + t) << 4) + lo;
#pragma unroll
        for (int r = 0; r < 8; ++r) Cb[(size_t)r * Nout] = acc[t][r];
    }
}

// ------------------------------------------------------- attention pipeline

__device__ __forceinline__ void edge_ep(const int* __restrict__ srcp, const int* __restrict__ dstp,
                                        int E0, int e, int& s, int& d) {
    if (e < E0) { s = srcp[e]; d = dstp[e]; }
    else        { s = d = e - E0; }            // self-loops appended at the end
}

__device__ __forceinline__ unsigned f2ord(float f) {
    unsigned u = __float_as_uint(f);
    return (u & 0x80000000u) ? ~u : (u | 0x80000000u);
}
__device__ __forceinline__ float ord2f(unsigned u) {
    return (u & 0x80000000u) ? __uint_as_float(u ^ 0x80000000u) : __uint_as_float(~u);
}

// s_src[n,h] = sum_c h[n,h,c]*a_src[h,c] ; same for s_dst (float4 inner loop)
__global__ void k_attn_scores(const float* __restrict__ h, const float* __restrict__ a_src,
                              const float* __restrict__ a_dst, float* __restrict__ s_src,
                              float* __restrict__ s_dst, int n, int H, int C) {
    int i = blockIdx.x * blockDim.x + threadIdx.x;
    if (i >= n * H) return;
    int node = i / H, hh = i % H;
    const float4* hp = (const float4*)(h + (size_t)node * H * C + hh * C);
    const float4* as = (const float4*)(a_src + hh * C);
    const float4* ad = (const float4*)(a_dst + hh * C);
    float ss = 0.f, sd = 0.f;
    for (int c = 0; c < (C >> 2); ++c) {
        float4 v = hp[c], a = as[c], b = ad[c];
        ss += v.x * a.x + v.y * a.y + v.z * a.z + v.w * a.w;
        sd += v.x * b.x + v.y * b.y + v.z * b.z + v.w * b.w;
    }
    s_src[i] = ss; s_dst[i] = sd;
}

// e = leaky_relu(s_src[src]+s_dst[dst]); segment max over dst via ordered-uint atomicMax
__global__ void k_edge_logits_max(const int* srcp, const int* dstp, int E0, int E,
                                  const float* __restrict__ s_src, const float* __restrict__ s_dst,
                                  float* __restrict__ e_buf, unsigned* __restrict__ umax, int H) {
    int i = blockIdx.x * blockDim.x + threadIdx.x;
    if (i >= E * H) return;
    int e = i / H, hh = i % H;
    int s, d; edge_ep(srcp, dstp, E0, e, s, d);
    float v = s_src[s * H + hh] + s_dst[d * H + hh];
    v = (v > 0.f) ? v : NEG_SLOPE * v;
    e_buf[i] = v;
    atomicMax(&umax[d * H + hh], f2ord(v));
}

// ex = exp(e - max[dst]); denom[dst] += ex
__global__ void k_edge_exp_sum(const int* srcp, const int* dstp, int E0, int E,
                               float* __restrict__ e_buf, const unsigned* __restrict__ umax,
                               float* __restrict__ denom, int H) {
    int i = blockIdx.x * blockDim.x + threadIdx.x;
    if (i >= E * H) return;
    int e = i / H, hh = i % H;
    int s, d; edge_ep(srcp, dstp, E0, e, s, d);
    float m = ord2f(umax[d * H + hh]);
    float ex = __expf(e_buf[i] - m);
    e_buf[i] = ex;
    atomicAdd(&denom[d * H + hh], ex);
}

// acc[dst,h,c] += (ex/denom[dst,h]) * hfeat[src,h,c]
__global__ void k_edge_scatter(const int* srcp, const int* dstp, int E0, int E,
                               const float* __restrict__ e_buf, const float* __restrict__ denom,
                               const float* __restrict__ hfeat, float* __restrict__ acc,
                               int H, int C) {
    long long total = (long long)E * H * C;
    long long i = (long long)blockIdx.x * blockDim.x + threadIdx.x;
    if (i >= total) return;
    int hc = H * C;
    int e = (int)(i / hc);
    int r = (int)(i % hc);
    int hh = r / C;
    int s, d; edge_ep(srcp, dstp, E0, e, s, d);
    float alpha = e_buf[(size_t)e * H + hh] / denom[d * H + hh];
    float val = hfeat[(size_t)s * hc + r] * alpha;
    atomicAdd(&acc[(size_t)d * hc + r], val);
}

__global__ void k_bias_act(const float* __restrict__ acc, const float* __restrict__ b,
                           float* __restrict__ out, int n, int F, int doRelu) {
    int i = blockIdx.x * blockDim.x + threadIdx.x;
    if (i >= n * F) return;
    float v = acc[i] + b[i % F];
    if (doRelu) v = (v > 0.f) ? v : 0.f;
    out[i] = v;
}

// ---------------------------------------------------------------- launcher

static inline int cdiv(long long a, long long b) { return (int)((a + b - 1) / b); }

extern "C" void kernel_launch(void* const* d_in, const int* in_sizes, int n_in,
                              void* d_out, int out_size, void* d_ws, size_t ws_size,
                              hipStream_t stream) {
    const float* x       = (const float*)d_in[0];
    const int*   eidx    = (const int*)d_in[1];
    const float* W1      = (const float*)d_in[2];
    const float* a_src1  = (const float*)d_in[3];
    const float* a_dst1  = (const float*)d_in[4];
    const float* b1      = (const float*)d_in[5];
    const float* W2      = (const float*)d_in[6];
    const float* a_src2  = (const float*)d_in[7];
    const float* a_dst2  = (const float*)d_in[8];
    const float* b2      = (const float*)d_in[9];
    const float* W3      = (const float*)d_in[10];
    const float* a_src3  = (const float*)d_in[11];
    const float* a_dst3  = (const float*)d_in[12];
    const float* b3      = (const float*)d_in[13];

    const int N  = in_sizes[0] / 3;     // 50000 nodes (in_ch = 3), N % 16 == 0
    const int E0 = in_sizes[1] / 2;     // 400000 real edges
    const int E  = E0 + N;              // + self loops
    const int* srcp = eidx;
    const int* dstp = eidx + E0;
    const int F = 256;                  // hidden feature width (4 heads * 64)

    // ---- carve workspace
    char* ws = (char*)d_ws;
    auto take = [&](size_t bytes) { char* p = ws; ws += (bytes + 255) & ~(size_t)255; return p; };
    float*     bufA  = (float*)take((size_t)N * F * sizeof(float));      // GEMM output h
    float*     bufB  = (float*)take((size_t)N * F * sizeof(float));      // scatter accumulator
    float*     bufC  = (float*)take((size_t)N * F * sizeof(float));      // activated layer output
    _Float16*  hf16  = (_Float16*)take((size_t)N * F * sizeof(_Float16));
    _Float16*  wf16  = (_Float16*)take((size_t)F * F * sizeof(_Float16)); // packed B
    float*     s_src = (float*)take((size_t)N * 4 * sizeof(float));
    float*     s_dst = (float*)take((size_t)N * 4 * sizeof(float));
    unsigned*  umax  = (unsigned*)take((size_t)N * 4 * sizeof(unsigned));
    float*     denom = (float*)take((size_t)N * 4 * sizeof(float));
    float*     e_buf = (float*)take((size_t)E * 4 * sizeof(float));
    (void)ws_size; (void)n_in; (void)out_size;

    const int T = 256;

    auto edge_phase = [&](const float* hfeat, const float* as, const float* ad,
                          float* acc, int H, int C) {
        k_attn_scores<<<cdiv((long long)N * H, T), T, 0, stream>>>(hfeat, as, ad, s_src, s_dst, N, H, C);
        k_fill_u32<<<cdiv((long long)N * H, T), T, 0, stream>>>(umax, 0u, N * H);
        k_fill_f32<<<cdiv((long long)N * H, T), T, 0, stream>>>(denom, 0.f, N * H);
        k_fill_f32<<<cdiv((long long)N * H * C, T), T, 0, stream>>>(acc, 0.f, N * H * C);
        k_edge_logits_max<<<cdiv((long long)E * H, T), T, 0, stream>>>(srcp, dstp, E0, E, s_src, s_dst, e_buf, umax, H);
        k_edge_exp_sum<<<cdiv((long long)E * H, T), T, 0, stream>>>(srcp, dstp, E0, E, e_buf, umax, denom, H);
        k_edge_scatter<<<cdiv((long long)E * H * C, T), T, 0, stream>>>(srcp, dstp, E0, E, e_buf, denom, hfeat, acc, H, C);
    };

    // ---------------- Layer 1: [N,3] @ [3,256] (K too small for WMMA), H=4,C=64
    k_gemm_small_k<<<cdiv((long long)N * F, T), T, 0, stream>>>(x, W1, bufA, N, 3, F);
    edge_phase(bufA, a_src1, a_dst1, bufB, 4, 64);
    k_bias_act<<<cdiv((long long)N * F, T), T, 0, stream>>>(bufB, b1, bufC, N, F, 1);

    // ---------------- Layer 2: [N,256] @ [256,256] via WMMA f16 (NT=4 strips)
    k_f32_to_f16_v4<<<cdiv((long long)N * F / 4, T), T, 0, stream>>>(bufC, hf16, N * F / 4);
    k_pack_B<<<cdiv((long long)F * F, T), T, 0, stream>>>(W2, wf16, F, F, 4);
    {
        int waves = (N / 16) * (F / 64);          // 3125 * 4 = 12500 strips
        k_wmma_gemm_f16<4><<<cdiv(waves, 8), 8 * 32, 0, stream>>>(hf16, wf16, bufA, F, F, waves);
    }
    edge_phase(bufA, a_src2, a_dst2, bufB, 4, 64);
    k_bias_act<<<cdiv((long long)N * F, T), T, 0, stream>>>(bufB, b2, bufC, N, F, 1);

    // ---------------- Layer 3: [N,256] @ [256,32] via WMMA f16 (NT=2), H=1,C=32
    k_f32_to_f16_v4<<<cdiv((long long)N * F / 4, T), T, 0, stream>>>(bufC, hf16, N * F / 4);
    k_pack_B<<<cdiv((long long)F * 32, T), T, 0, stream>>>(W3, wf16, F, 32, 2);
    {
        int waves = (N / 16) * 1;                 // 3125 strips of 16x32
        k_wmma_gemm_f16<2><<<cdiv(waves, 8), 8 * 32, 0, stream>>>(hf16, wf16, bufA, F, 32, waves);
    }
    edge_phase(bufA, a_src3, a_dst3, bufB, 1, 32);
    k_bias_act<<<cdiv((long long)N * 32, T), T, 0, stream>>>(bufB, b3, (float*)d_out, N, 32, 0);
}
